// GatedAttentionPooling_4398046511884
// MI455X (gfx1250) — matrix operand order repaired and checked by reading, not compiled, not run on hardware
//
#include <hip/hip_runtime.h>
#include <hip/hip_bf16.h>

typedef __attribute__((ext_vector_type(16))) __bf16  v16bf;
typedef __attribute__((ext_vector_type(8)))  float   v8f;
typedef __attribute__((ext_vector_type(8)))  unsigned v8u;

static __device__ __forceinline__ unsigned short f2bf(float f) {
  unsigned u = __float_as_uint(f);
  unsigned r = u + 0x7FFFu + ((u >> 16) & 1u);   // round-to-nearest-even
  return (unsigned short)(r >> 16);
}
static __device__ __forceinline__ unsigned pack2bf(float lo, float hi) {
  return (unsigned)f2bf(lo) | ((unsigned)f2bf(hi) << 16);
}
static __device__ __forceinline__ v16bf as_bf(v8u x) {
  union { v8u u; v16bf b; } t; t.u = x; return t.b;
}
// A-fragment: dwords {0..3} and {8..11} from a 16B-aligned base -> 2x b128
static __device__ __forceinline__ v8u frag_split(const unsigned* base) {
  const uint4* p = (const uint4*)base;
  uint4 lo = p[0], hi = p[2];
  v8u r;
  r[0] = lo.x; r[1] = lo.y; r[2] = lo.z; r[3] = lo.w;
  r[4] = hi.x; r[5] = hi.y; r[6] = hi.z; r[7] = hi.w;
  return r;
}
// B-fragment: 8 consecutive dwords from a 16B-aligned base -> 2x b128
static __device__ __forceinline__ v8u frag_consec(const unsigned* base) {
  const uint4* p = (const uint4*)base;
  uint4 lo = p[0], hi = p[1];
  v8u r;
  r[0] = lo.x; r[1] = lo.y; r[2] = lo.z; r[3] = lo.w;
  r[4] = hi.x; r[5] = hi.y; r[6] = hi.z; r[7] = hi.w;
  return r;
}
// monotone order-preserving f32 <-> u32 map for atomicMax
static __device__ __forceinline__ unsigned order_f32(float f) {
  unsigned u = __float_as_uint(f);
  return (u & 0x80000000u) ? ~u : (u | 0x80000000u);
}
static __device__ __forceinline__ float unorder_f32(unsigned k) {
  unsigned u = (k & 0x80000000u) ? (k & 0x7FFFFFFFu) : ~k;
  return __uint_as_float(u);
}

// ---------------- init: zero Z, seg_sum, seg_max keys ----------------
__global__ void __launch_bounds__(256)
init_kernel(float* __restrict__ Z, int zn,
            float* __restrict__ segsum, unsigned* __restrict__ segkey, int S) {
  int i = blockIdx.x * 256 + threadIdx.x;
  if (i < zn) Z[i] = 0.0f;
  if (i < S) { segsum[i] = 0.0f; segkey[i] = 0u; }
}

// ---------------- prep: W1,W2 -> N-major bf16 ----------------
__global__ void __launch_bounds__(256)
prep_kernel(const float* __restrict__ W1, const float* __restrict__ W2,
            unsigned short* __restrict__ w1t, unsigned short* __restrict__ w2t) {
  int i = blockIdx.x * 256 + threadIdx.x;
  if (i < 256 * 128) {                // W1 [256][128] -> w1t [128][256]
    int n = i & 127, k = i >> 7;
    w1t[n * 256 + k] = f2bf(W1[k * 128 + n]);
  }
  if (i < 128 * 64) {                 // W2 [128][64] -> w2t [64][128]
    int n = i & 63, k = i >> 6;
    w2t[n * 128 + k] = f2bf(W2[k * 64 + n]);
  }
}

// ---------------- logits: fused 3-layer MLP via bf16 WMMA ----------------
// block = 256 threads = 8 waves; each wave owns a 16-row tile (128 rows/block)
__global__ void __launch_bounds__(256)
logits_kernel(const float* __restrict__ src,
              const unsigned* __restrict__ w1t,   // bf16 pairs, [128][128] dwords
              const unsigned* __restrict__ w2t,   // bf16 pairs, [64][64] dwords
              const float* __restrict__ W3,
              const float* __restrict__ b1, const float* __restrict__ b2,
              const float* __restrict__ b3,
              float* __restrict__ logits, int N) {
  extern __shared__ char smem[];
  unsigned*       srcS = (unsigned*)smem;                 // [128][128] bf16 pairs (64 KB)
  unsigned short* h1s  = (unsigned short*)(smem + 65536); // 8 waves x [16][128] bf16 (32 KB)
  unsigned*       h1u  = (unsigned*)(smem + 65536);

  const int tid  = threadIdx.x;
  const int base = blockIdx.x * 128;

  // stage src tile as bf16: float4 global reads, uint2 LDS writes
  {
    uint2* srcS2 = (uint2*)srcS;                   // [128][64] dword pairs
    for (int i = tid; i < 128 * 64; i += 256) {
      int r = i >> 6, q = i & 63;
      int grow = base + r;
      float4 v = make_float4(0.f, 0.f, 0.f, 0.f);
      if (grow < N) v = ((const float4*)(src + (size_t)grow * 256))[q];
      srcS2[i] = make_uint2(pack2bf(v.x, v.y), pack2bf(v.z, v.w));
    }
  }
  __syncthreads();

  const int wave = tid >> 5;
  const int lane = tid & 31;
  const int l16  = lane & 15;
  const int hi16 = lane >> 4;        // 0: lanes 0-15, 1: lanes 16-31
  const int m0   = wave * 16;

  // per-lane fragment base offsets (dwords)
  const unsigned* aBase1 = srcS + (m0 + l16) * 128 + hi16 * 8;        // + kt*16
  const unsigned* aBase2 = h1u + wave * 1024 + l16 * 64 + hi16 * 8;   // + kt*16

  // ---- layer 1: relu(src[16x256] @ W1[256x128] + b1) -> h1 bf16 in LDS ----
  #pragma unroll 1
  for (int nt = 0; nt < 8; ++nt) {
    float bias = b1[nt * 16 + l16];
    v8f acc;
    #pragma unroll
    for (int j = 0; j < 8; ++j) acc[j] = bias;
    const unsigned* bBase = w1t + (nt * 16 + l16) * 128 + hi16 * 8;   // + kt*16
    #pragma unroll
    for (int kt = 0; kt < 8; ++kt) {
      v8u a = frag_consec(aBase1 + kt * 16);
      v8u b = frag_consec(bBase + kt * 16);
      acc = __builtin_amdgcn_wmma_f32_16x16x32_bf16(false, as_bf(a), false, as_bf(b),
                                                    (short)0, acc, false, false);
    }
    #pragma unroll
    for (int j = 0; j < 8; ++j) {
      float v = acc[j] > 0.0f ? acc[j] : 0.0f;    // ReLU
      int mloc = j + 8 * hi16;                    // C layout: M = vgpr + 8*half
      h1s[wave * 2048 + mloc * 128 + nt * 16 + l16] = f2bf(v);
    }
  }
  // h1 region is private to this wave; LDS ops stay in-order per wave.

  // ---- layer 2 (relu(h1 @ W2 + b2)) fused with layer 3 dot (W3) ----
  float part[8];
  #pragma unroll
  for (int j = 0; j < 8; ++j) part[j] = 0.0f;

  #pragma unroll 1
  for (int nt = 0; nt < 4; ++nt) {
    float bias = b2[nt * 16 + l16];
    v8f acc;
    #pragma unroll
    for (int j = 0; j < 8; ++j) acc[j] = bias;
    const unsigned* bBase = w2t + (nt * 16 + l16) * 64 + hi16 * 8;    // + kt*16
    #pragma unroll
    for (int kt = 0; kt < 4; ++kt) {
      v8u a = frag_consec(aBase2 + kt * 16);
      v8u b = frag_consec(bBase + kt * 16);
      acc = __builtin_amdgcn_wmma_f32_16x16x32_bf16(false, as_bf(a), false, as_bf(b),
                                                    (short)0, acc, false, false);
    }
    float w3v = W3[nt * 16 + l16];
    #pragma unroll
    for (int j = 0; j < 8; ++j) {
      float v = acc[j] > 0.0f ? acc[j] : 0.0f;    // ReLU
      part[j] += v * w3v;                         // partial dot with W3 column
    }
  }

  // reduce each row's partial dot across its 16-lane half (wave32 shuffles)
  #pragma unroll
  for (int j = 0; j < 8; ++j) {
    float s = part[j];
    s += __shfl_xor(s, 1, 32);
    s += __shfl_xor(s, 2, 32);
    s += __shfl_xor(s, 4, 32);
    s += __shfl_xor(s, 8, 32);
    part[j] = s;
  }
  float bb3 = b3[0];
  if (l16 == 0) {
    #pragma unroll
    for (int j = 0; j < 8; ++j) {
      int grow = base + m0 + j + 8 * hi16;
      if (grow < N) logits[grow] = part[j] + bb3;
    }
  }
}

// ---------------- segment max (atomic on ordered-uint key) ----------------
__global__ void __launch_bounds__(256)
segmax_kernel(const float* __restrict__ logits, const int* __restrict__ index,
              unsigned* __restrict__ segkey, int N) {
  int i = blockIdx.x * 256 + threadIdx.x;
  if (i < N) atomicMax(&segkey[index[i]], order_f32(logits[i]));
}

// ---------------- exp + segment sum ----------------
__global__ void __launch_bounds__(256)
exsum_kernel(const float* __restrict__ logits, const int* __restrict__ index,
             const unsigned* __restrict__ segkey,
             float* __restrict__ ex, float* __restrict__ segsum, int N) {
  int i = blockIdx.x * 256 + threadIdx.x;
  if (i < N) {
    int g = index[i];
    unsigned k = segkey[g];
    float m = (k == 0u) ? 0.0f : unorder_f32(k);   // empty-segment guard like reference
    float e = __expf(logits[i] - m);
    ex[i] = e;
    unsafeAtomicAdd(&segsum[g], e);                // global_atomic_add_f32
  }
}

// ---------------- weighted scatter-add: Z[g] += att*src[i] ----------------
// one row per 32-lane wave: lane handles 8 columns (2x float4 loads)
__global__ void __launch_bounds__(256)
scatter_kernel(const float* __restrict__ src, const int* __restrict__ index,
               const float* __restrict__ ex, const float* __restrict__ segsum,
               float* __restrict__ Z, int N) {
  int row  = blockIdx.x * 8 + (threadIdx.x >> 5);
  int lane = threadIdx.x & 31;
  if (row < N) {
    int g = index[row];
    float att = ex[row] / segsum[g];
    const float4* s = (const float4*)(src + (size_t)row * 256);
    float* zp = Z + (size_t)g * 256;
    float4 v0 = s[lane];
    float4 v1 = s[lane + 32];
    unsafeAtomicAdd(&zp[lane * 4 + 0], att * v0.x);
    unsafeAtomicAdd(&zp[lane * 4 + 1], att * v0.y);
    unsafeAtomicAdd(&zp[lane * 4 + 2], att * v0.z);
    unsafeAtomicAdd(&zp[lane * 4 + 3], att * v0.w);
    unsafeAtomicAdd(&zp[128 + lane * 4 + 0], att * v1.x);
    unsafeAtomicAdd(&zp[128 + lane * 4 + 1], att * v1.y);
    unsafeAtomicAdd(&zp[128 + lane * 4 + 2], att * v1.z);
    unsafeAtomicAdd(&zp[128 + lane * 4 + 3], att * v1.w);
  }
}

extern "C" void kernel_launch(void* const* d_in, const int* in_sizes, int n_in,
                              void* d_out, int out_size, void* d_ws, size_t ws_size,
                              hipStream_t stream) {
  const float* src   = (const float*)d_in[0];
  const int*   index = (const int*)d_in[1];
  // d_in[2]=dim, d_in[3]=dim_size (scalars; constants below match reference)
  const float* W1 = (const float*)d_in[4];
  const float* b1 = (const float*)d_in[5];
  const float* W2 = (const float*)d_in[6];
  const float* b2 = (const float*)d_in[7];
  const float* W3 = (const float*)d_in[8];
  const float* b3 = (const float*)d_in[9];

  const int N = in_sizes[0] / 256;   // DIM_IN = 256
  const int S = 50000;               // DIM_SIZE

  char* ws = (char*)d_ws;
  float*          logits = (float*)ws;          ws += (size_t)N * 4;
  float*          ex     = (float*)ws;          ws += (size_t)N * 4;
  unsigned*       segkey = (unsigned*)ws;       ws += (size_t)S * 4;
  float*          segsum = (float*)ws;          ws += (size_t)S * 4;
  unsigned short* w1t    = (unsigned short*)ws; ws += 256 * 128 * 2;
  unsigned short* w2t    = (unsigned short*)ws;
  float* Z = (float*)d_out;

  const int smem_bytes = 65536 + 32768;         // src tile + h1 staging
  hipFuncSetAttribute((const void*)logits_kernel,
                      hipFuncAttributeMaxDynamicSharedMemorySize, smem_bytes);

  int zn = S * 256;
  init_kernel<<<(zn + 255) / 256, 256, 0, stream>>>(Z, zn, segsum, segkey, S);
  prep_kernel<<<(256 * 128 + 255) / 256, 256, 0, stream>>>(W1, W2, w1t, w2t);
  logits_kernel<<<(N + 127) / 128, 256, smem_bytes, stream>>>(
      src, (const unsigned*)w1t, (const unsigned*)w2t, W3, b1, b2, b3, logits, N);
  segmax_kernel<<<(N + 255) / 256, 256, 0, stream>>>(logits, index, segkey, N);
  exsum_kernel<<<(N + 255) / 256, 256, 0, stream>>>(logits, index, segkey, ex, segsum, N);
  scatter_kernel<<<(N + 7) / 8, 256, 0, stream>>>(src, index, ex, segsum, Z, N);
}
